// PointNet2_model_29738353557550
// MI455X (gfx1250) — matrix operand order repaired and checked by reading, not compiled, not run on hardware
//
#include <hip/hip_runtime.h>

// ---------------------------------------------------------------------------
// PointNet++ (2x set abstraction) for MI455X / gfx1250, wave32 + WMMA +
// async global->LDS gathers. Structure: prep (pack weights to WMMA-B fragment
// order with [feat|xyz|pad] K permutation, fold BN, transpose descriptors to
// sample-major f16) -> FPS -> fused ball-query + async gather + 3x WMMA MLP
// + maxpool per center -> FPS2 -> fused SA2 -> final gathers.
// ---------------------------------------------------------------------------

typedef __attribute__((ext_vector_type(16))) _Float16 v16h;
typedef __attribute__((ext_vector_type(8)))  float    v8f;

union FragU { unsigned u[8]; v16h h; };

// ---------------------------------------------------------------------------
// Prep: pack W (cin x cout, f32) into f16 WMMA B fragments, K padded to 32.
// Packed-K order for layer 1 (cfeat >= 0): kk<cfeat -> feature channel kk
// (orig row kk+3); cfeat<=kk<cfeat+3 -> xyz (orig row kk-cfeat); else zero.
// cfeat < 0 -> identity. Fragment layout: block (kt,nt) -> 1024 halves,
// half = lane*16 + i*2 + e at kk = kt*32+(i&3)*2+(i>>2)*16+(lane>>4)*8+e,
// col = nt*16 + (lane&15). BN folded: scale=g*rsqrt(var+eps),
// shift=(b-mu)*scale+beta.
// ---------------------------------------------------------------------------
__global__ __launch_bounds__(256) void prep_layer(
    const float* __restrict__ W, const float* __restrict__ b,
    const float* __restrict__ g, const float* __restrict__ beta,
    const float* __restrict__ mu, const float* __restrict__ var,
    int cin, int cout, int KT, int NT, int cfeat,
    _Float16* __restrict__ Wdst, float* __restrict__ scale,
    float* __restrict__ shift)
{
    const int stride = gridDim.x * blockDim.x;
    const int total  = KT * NT * 1024;
    for (int idx = blockIdx.x * blockDim.x + threadIdx.x; idx < total; idx += stride) {
        int blk  = idx >> 10;
        int h    = idx & 1023;
        int lane = h >> 4;
        int jj   = h & 15;
        int i    = jj >> 1;
        int e    = jj & 1;
        int kt   = blk / NT;
        int nt   = blk - kt * NT;
        int kk   = kt * 32 + ((i & 3) << 1) + ((i >> 2) << 4) + ((lane >> 4) << 3) + e;
        int col  = nt * 16 + (lane & 15);
        int orig = (cfeat < 0) ? kk : ((kk < cfeat) ? (kk + 3) : (kk - cfeat));
        float v  = (kk < cin) ? W[(size_t)orig * cout + col] : 0.0f;
        Wdst[idx] = (_Float16)v;
    }
    for (int c = blockIdx.x * blockDim.x + threadIdx.x; c < cout; c += stride) {
        float a  = g[c] * rsqrtf(var[c] + 1e-5f);
        scale[c] = a;
        shift[c] = (b[c] - mu[c]) * a + beta[c];
    }
}

// Transpose + convert descriptors (128 x N ch-major f32) -> (N x 128 f16).
__global__ __launch_bounds__(256) void pack_desc_f16(
    const float* __restrict__ desc, int n, _Float16* __restrict__ out)
{
    const int stride = gridDim.x * blockDim.x;
    const int total  = n * 128;
    for (int idx = blockIdx.x * blockDim.x + threadIdx.x; idx < total; idx += stride) {
        int j = idx >> 7;
        int c = idx & 127;
        out[idx] = (_Float16)desc[(size_t)c * n + j];
    }
}

// ---------------------------------------------------------------------------
// Farthest point sampling: single block, sequential over npoint iterations.
// ---------------------------------------------------------------------------
__global__ __launch_bounds__(1024) void fps_kernel(
    const float* __restrict__ pts, int n, int npoint,
    int* __restrict__ outIdx, float* __restrict__ outXyz)
{
    __shared__ float dist[6016];
    __shared__ float sval[1024];
    __shared__ int   sidx[1024];
    const int tid = threadIdx.x;

    for (int j = tid; j < n; j += 1024) dist[j] = 1e10f;
    __syncthreads();

    int far = 0;
    for (int it = 0; it < npoint; ++it) {
        if (tid == 0) {
            outIdx[it] = far;
            outXyz[it * 3 + 0] = pts[far * 3 + 0];
            outXyz[it * 3 + 1] = pts[far * 3 + 1];
            outXyz[it * 3 + 2] = pts[far * 3 + 2];
        }
        const float fx = pts[far * 3 + 0];
        const float fy = pts[far * 3 + 1];
        const float fz = pts[far * 3 + 2];

        float bv = -1.0f; int bi = 0;
        for (int j = tid; j < n; j += 1024) {
            float dx = pts[j * 3 + 0] - fx;
            float dy = pts[j * 3 + 1] - fy;
            float dz = pts[j * 3 + 2] - fz;
            float d  = dx * dx + dy * dy + dz * dz;
            float nd = fminf(dist[j], d);
            dist[j]  = nd;
            if (nd > bv) { bv = nd; bi = j; }
        }
        sval[tid] = bv; sidx[tid] = bi;
        __syncthreads();
        for (int st = 512; st > 0; st >>= 1) {
            if (tid < st) {
                float ov = sval[tid + st]; int oi = sidx[tid + st];
                float mv = sval[tid];      int mi = sidx[tid];
                if (ov > mv || (ov == mv && oi < mi)) { sval[tid] = ov; sidx[tid] = oi; }
            }
            __syncthreads();
        }
        far = sidx[0];
        __syncthreads();
    }
}

// ---------------------------------------------------------------------------
// One fused MLP layer with 2xN register blocking: each A fragment feeds two
// WMMAs on adjacent N tiles (their packed B blocks are contiguous 2KB).
// ---------------------------------------------------------------------------
template <int MT, int NT, int KT>
__device__ inline void mlp_layer(const _Float16* __restrict__ src, int srcStride,
                                 _Float16* __restrict__ dst, int dstStride,
                                 const _Float16* __restrict__ Wp,
                                 const float* __restrict__ scale,
                                 const float* __restrict__ shift,
                                 int wave, int lane)
{
    constexpr int NPAIR = NT / 2;
    const int rowA  = (lane & 15);
    const int hiOff = ((lane >> 4) << 3);
    for (int t = wave; t < MT * NPAIR; t += 8) {
        int mt  = t % MT;
        int np  = t / MT;
        int nt0 = np * 2;
        v8f acc0 = {}, acc1 = {};
        const _Float16* aBase = src + (size_t)(mt * 16 + rowA) * srcStride + hiOff;
        for (int kt = 0; kt < KT; ++kt) {
            FragU a, b0, b1;
            const _Float16* ap = aBase + kt * 32;
#pragma unroll
            for (int i = 0; i < 4; ++i) {
                a.u[i]     = *(const unsigned*)(ap + 2 * i);
                a.u[4 + i] = *(const unsigned*)(ap + 16 + 2 * i);
            }
            const unsigned* bp =
                (const unsigned*)(Wp + ((size_t)(kt * NT + nt0) << 10) + (lane << 4));
            if (kt + 1 < KT)
                __builtin_prefetch((const void*)(Wp + ((size_t)((kt + 1) * NT + nt0) << 10) +
                                                 (lane << 4)), 0, 1);
#pragma unroll
            for (int i = 0; i < 8; ++i) b0.u[i] = bp[i];
#pragma unroll
            for (int i = 0; i < 8; ++i) b1.u[i] = bp[512 + i];   // next 1024-half block
            acc0 = __builtin_amdgcn_wmma_f32_16x16x32_f16(
                false, a.h, false, b0.h, (short)0, acc0, false, false);
            acc1 = __builtin_amdgcn_wmma_f32_16x16x32_f16(
                false, a.h, false, b1.h, (short)0, acc1, false, false);
        }
        const int mB = mt * 16 + ((lane >> 4) << 3);
#pragma unroll
        for (int half = 0; half < 2; ++half) {
            const int   ncol = (nt0 + half) * 16 + (lane & 15);
            const float a_s  = scale[ncol];
            const float b_s  = shift[ncol];
            const v8f&  acc  = half ? acc1 : acc0;
#pragma unroll
            for (int r = 0; r < 8; ++r) {
                float v = fmaxf(acc[r] * a_s + b_s, 0.0f);
                dst[(size_t)(mB + r) * dstStride + ncol] = (_Float16)v;
            }
        }
    }
}

// ---------------------------------------------------------------------------
// Fused set-abstraction kernel: one block (256 thr / 8 waves) per center.
// Ball query (wave 0) -> async global->LDS feature gather + VALU xyz ->
// 3 WMMA MLP layers -> maxpool.
// K layout of bufA rows: [feat(CF) | xyz(3) | pad] with stride KPAD.
// feat: sample-major f16, row stride CF halves (CF*2 bytes per row).
// ---------------------------------------------------------------------------
template <int NS, int CF, int KPAD, int C1, int C2, int C3,
          bool OUT_CH_MAJOR, bool OUT_F16>
__global__ __launch_bounds__(256) void sa_kernel(
    const float* __restrict__ srcXyz, int n,
    const float* __restrict__ centers,
    const _Float16* __restrict__ feat,
    float radius2,
    const _Float16* __restrict__ W1, const float* __restrict__ sc1, const float* __restrict__ sh1,
    const _Float16* __restrict__ W2, const float* __restrict__ sc2, const float* __restrict__ sh2,
    const _Float16* __restrict__ W3, const float* __restrict__ sc3, const float* __restrict__ sh3,
    void* __restrict__ outPts, int outLd)
{
    constexpr int MT  = NS / 16;
    constexpr int NT1 = C1 / 16, KT1 = KPAD / 32;
    constexpr int NT2 = C2 / 16, KT2 = C1 / 32;
    constexpr int NT3 = C3 / 16, KT3 = C2 / 32;
    constexpr int BPL = (CF * 2) / 32;   // feature bytes per lane (8 or 16)

    extern __shared__ char smem[];
    int*      nbr  = (int*)smem;
    _Float16* bufA = (_Float16*)(smem + ((NS * 4 + 15) & ~15));
    _Float16* bufB = bufA + (size_t)NS * KPAD;

    const int tid  = threadIdx.x;
    const int lane = tid & 31;
    const int wave = tid >> 5;
    const int s    = blockIdx.x;

    const float cx = centers[s * 3 + 0];
    const float cy = centers[s * 3 + 1];
    const float cz = centers[s * 3 + 2];

    // ---- ball query: first NS in-radius indices in ascending order ----
    if (wave == 0) {
        int count = 0;
        for (int base = 0; base < n && count < NS; base += 32) {
            int  j  = base + lane;
            bool in = false;
            if (j < n) {
                float dx = srcXyz[j * 3 + 0] - cx;
                float dy = srcXyz[j * 3 + 1] - cy;
                float dz = srcXyz[j * 3 + 2] - cz;
                in = (dx * dx + dy * dy + dz * dz) <= radius2;
            }
            unsigned mask = (unsigned)__ballot(in);
            int pos = count + __popc(mask & ((1u << lane) - 1u));
            if (in && pos < NS) nbr[pos] = j;
            count += __popc(mask);
        }
        if (count > NS) count = NS;
        if (count == 0) { if (lane == 0) nbr[0] = n - 1; count = 1; }
        int first = nbr[0];
        for (int p = count + lane; p < NS; p += 32) nbr[p] = first;
    }
    __syncthreads();

    // ---- async feature gather: one wave moves one sample row (CF*2 bytes,
    //      BPL bytes per lane) straight into LDS, ASYNCcnt-tracked ----
    {
        const unsigned ldsBase = (unsigned)(size_t)(void*)bufA;
        for (int i = wave; i < NS; i += 8) {
            int j = nbr[i];
            const void* src = (const char*)(feat + (size_t)j * CF) + lane * BPL;
            unsigned dst = ldsBase + (unsigned)(i * (KPAD * 2) + lane * BPL);
            if (BPL == 8) {
                asm volatile("global_load_async_to_lds_b64 %0, %1, off"
                             :: "v"(dst), "v"(src) : "memory");
            } else {
                asm volatile("global_load_async_to_lds_b128 %0, %1, off"
                             :: "v"(dst), "v"(src) : "memory");
            }
        }
    }

    // ---- xyz (centered) + zero pad for K slots [CF, KPAD) ----
    for (int idx = tid; idx < NS * (KPAD - CF); idx += 256) {
        int i = idx / (KPAD - CF);
        int k = idx - i * (KPAD - CF);
        float v = 0.0f;
        if (k < 3) {
            float c = (k == 0) ? cx : ((k == 1) ? cy : cz);
            v = srcXyz[nbr[i] * 3 + k] - c;
        }
        bufA[(size_t)i * KPAD + CF + k] = (_Float16)v;
    }
    asm volatile("s_wait_asynccnt 0x0" ::: "memory");
    __syncthreads();

    mlp_layer<MT, NT1, KT1>(bufA, KPAD, bufB, 256, W1, sc1, sh1, wave, lane);
    __syncthreads();
    mlp_layer<MT, NT2, KT2>(bufB, 256, bufA, KPAD, W2, sc2, sh2, wave, lane);
    __syncthreads();
    mlp_layer<MT, NT3, KT3>(bufA, KPAD, bufB, 256, W3, sc3, sh3, wave, lane);
    __syncthreads();

    // ---- max over samples; C3 == 256 == blockDim ----
    {
        int   c = tid;
        float m = 0.0f;  // post-ReLU values are >= 0
        for (int i = 0; i < NS; ++i)
            m = fmaxf(m, (float)bufB[(size_t)i * 256 + c]);
        if (OUT_F16) {
            _Float16* o = (_Float16*)outPts;
            if (OUT_CH_MAJOR) o[(size_t)c * outLd + s] = (_Float16)m;
            else              o[(size_t)s * outLd + c] = (_Float16)m;
        } else {
            float* o = (float*)outPts;
            if (OUT_CH_MAJOR) o[(size_t)c * outLd + s] = m;
            else              o[(size_t)s * outLd + c] = m;
        }
    }
}

// ---------------------------------------------------------------------------
// Final gathers. d_out: [0,768) l2_xyz; [768,66304) l2_pts (written by SA2);
// [66304,99072) desc(128x256); [99072,99328) scores; [99328,99584) fps_idx.
// ---------------------------------------------------------------------------
__global__ __launch_bounds__(256) void finalize_kernel(
    const int* __restrict__ fps1, const int* __restrict__ fps2,
    const float* __restrict__ xyz2,
    const float* __restrict__ descCoarse,  // 128 x 6000
    const float* __restrict__ scores,      // 6000
    float* __restrict__ out)
{
    const int tid = threadIdx.x;
    if (blockIdx.x == 0) {
        int g = fps1[fps2[tid]];
        out[99328 + tid] = (float)g;
        out[99072 + tid] = scores[g];
        for (int i = tid; i < 768; i += 256) out[i] = xyz2[i];
    } else {
        int c = blockIdx.x - 1;   // 0..127
        int j = tid;
        int g = fps1[fps2[j]];
        out[66304 + c * 256 + j] = descCoarse[(size_t)c * 6000 + g];
    }
}

// ---------------------------------------------------------------------------
// Host launcher
// ---------------------------------------------------------------------------
extern "C" void kernel_launch(void* const* d_in, const int* in_sizes, int n_in,
                              void* d_out, int out_size, void* d_ws, size_t ws_size,
                              hipStream_t stream)
{
    (void)in_sizes; (void)n_in; (void)out_size; (void)ws_size;

    const float* kp     = (const float*)d_in[0];  // (1,6000,3)
    const float* desc   = (const float*)d_in[1];  // (1,128,6000)
    const float* descC  = (const float*)d_in[2];  // (1,128,6000)
    const float* scores = (const float*)d_in[3];  // (1,6000,1)
    float*       out    = (float*)d_out;

    // ---- workspace carve (256B aligned chunks) ----
    char* p = (char*)d_ws;
    auto alloc = [&](size_t bytes) -> void* {
        void* r = (void*)p;
        p += (bytes + 255) & ~(size_t)255;
        return r;
    };
    int*      fps1    = (int*)     alloc(5000 * sizeof(int));
    float*    xyz1    = (float*)   alloc(5000 * 3 * sizeof(float));
    _Float16* l1ptsH  = (_Float16*)alloc((size_t)5000 * 256 * sizeof(_Float16));
    int*      fps2    = (int*)     alloc(256 * sizeof(int));
    float*    xyz2    = (float*)   alloc(256 * 3 * sizeof(float));
    _Float16* desc16T = (_Float16*)alloc((size_t)6000 * 128 * sizeof(_Float16));

    const int cinA[6]   = {131, 128, 128, 259, 256, 256};
    const int coutA[6]  = {128, 128, 256, 256, 256, 256};
    const int KTA[6]    = {5, 4, 4, 9, 8, 8};       // KPAD/32
    const int NTA[6]    = {8, 8, 16, 16, 16, 16};   // cout/16
    const int cfeatA[6] = {128, -1, -1, 256, -1, -1};

    _Float16* Wp[6]; float* SC[6]; float* SH[6];
    for (int l = 0; l < 6; ++l) {
        Wp[l] = (_Float16*)alloc((size_t)KTA[l] * NTA[l] * 1024 * sizeof(_Float16));
        SC[l] = (float*)alloc(coutA[l] * sizeof(float));
        SH[l] = (float*)alloc(coutA[l] * sizeof(float));
    }

    // ---- prep: pack weights + fold BN; transpose descriptors to f16 ----
    for (int l = 0; l < 6; ++l) {
        const float* W    = (const float*)d_in[4 + l * 6 + 0];
        const float* b    = (const float*)d_in[4 + l * 6 + 1];
        const float* g    = (const float*)d_in[4 + l * 6 + 2];
        const float* beta = (const float*)d_in[4 + l * 6 + 3];
        const float* mu   = (const float*)d_in[4 + l * 6 + 4];
        const float* var  = (const float*)d_in[4 + l * 6 + 5];
        prep_layer<<<256, 256, 0, stream>>>(W, b, g, beta, mu, var,
                                            cinA[l], coutA[l], KTA[l], NTA[l], cfeatA[l],
                                            Wp[l], SC[l], SH[l]);
    }
    pack_desc_f16<<<1024, 256, 0, stream>>>(desc, 6000, desc16T);

    // ---- SA1 ----
    fps_kernel<<<1, 1024, 0, stream>>>(kp, 6000, 5000, fps1, xyz1);
    {
        const size_t shm = ((256 * 4 + 15) & ~15) +
                           (size_t)256 * 160 * 2 + (size_t)256 * 256 * 2;  // ~214 KB
        sa_kernel<256, 128, 160, 128, 128, 256, false, true>
            <<<5000, 256, shm, stream>>>(
                kp, 6000, xyz1, desc16T, 0.2f * 0.2f,
                Wp[0], SC[0], SH[0], Wp[1], SC[1], SH[1], Wp[2], SC[2], SH[2],
                (void*)l1ptsH, 256);
    }

    // ---- SA2 ----
    fps_kernel<<<1, 1024, 0, stream>>>(xyz1, 5000, 256, fps2, xyz2);
    {
        const size_t shm = ((128 * 4 + 15) & ~15) +
                           (size_t)128 * 288 * 2 + (size_t)128 * 256 * 2;  // ~140 KB
        sa_kernel<128, 256, 288, 256, 256, 256, true, false>
            <<<256, 256, shm, stream>>>(
                xyz1, 5000, xyz2, l1ptsH, 0.4f * 0.4f,
                Wp[3], SC[3], SH[3], Wp[4], SC[4], SH[4], Wp[5], SC[5], SH[5],
                (void*)(out + 768), 256);
    }

    // ---- outputs ----
    finalize_kernel<<<129, 256, 0, stream>>>(fps1, fps2, xyz2, descC, scores, out);
}
